// CrossAttention_19645180412127
// MI455X (gfx1250) — matrix operand order repaired and checked by reading, not compiled, not run on hardware
//
#include <hip/hip_runtime.h>

typedef _Float16 half_t;
typedef __attribute__((ext_vector_type(16))) _Float16 v16h;
typedef __attribute__((ext_vector_type(8)))  _Float16 v8h;
typedef __attribute__((ext_vector_type(8)))  float    v8f;

#define DMODEL 128
#define KTILE  64      // keys per iteration
#define NQTILE 128     // query rows per workgroup (8 waves x 16 rows)

__device__ __forceinline__ void wait_async0() {
#if __has_builtin(__builtin_amdgcn_s_wait_asynccnt)
  __builtin_amdgcn_s_wait_asynccnt(0);
#else
  asm volatile("s_wait_asynccnt 0x0" ::: "memory");
#endif
}

// ---------------------------------------------------------------------------
// Projection: Y[n][e] = sum_d X[n][d] * W[e][d] + b[e], output f16.
// Block: 256 threads = 8 waves; each wave one 16x16 tile, K=128 in 4 WMMAs.
// transposeN == 0 : Y row-major [row][e]
// transposeN == N : Y transposed per batch [b][e][n]  (for V)
// ---------------------------------------------------------------------------
__global__ __launch_bounds__(256) void proj_kernel(
    const float* __restrict__ X, const float* __restrict__ W,
    const float* __restrict__ bias, half_t* __restrict__ Y, int transposeN) {
  const int tid  = threadIdx.x;
  const int wave = tid >> 5;
  const int lane = tid & 31;
  const int n16  = lane & 15;
  const int g    = lane >> 4;
  const int rowbase = blockIdx.x * 16;
  const int ebase   = wave * 16;

  v8f acc = {};
#pragma unroll
  for (int c = 0; c < 4; ++c) {
    // A fragment: halves 0..15 -> k = 16g+h
    const float* xr = X + (size_t)(rowbase + n16) * DMODEL + c * 32 + 16 * g;
    float4 x0 = *(const float4*)(xr);
    float4 x1 = *(const float4*)(xr + 4);
    float4 x2 = *(const float4*)(xr + 8);
    float4 x3 = *(const float4*)(xr + 12);
    v16h a;
    a[0]=(half_t)x0.x; a[1]=(half_t)x0.y; a[2]=(half_t)x0.z; a[3]=(half_t)x0.w;
    a[4]=(half_t)x1.x; a[5]=(half_t)x1.y; a[6]=(half_t)x1.z; a[7]=(half_t)x1.w;
    a[8]=(half_t)x2.x; a[9]=(half_t)x2.y; a[10]=(half_t)x2.z; a[11]=(half_t)x2.w;
    a[12]=(half_t)x3.x; a[13]=(half_t)x3.y; a[14]=(half_t)x3.z; a[15]=(half_t)x3.w;

    // B fragment: B[k][n] = W[ebase+n][c*32 + 16g + h], h contiguous
    const float* wr = W + (size_t)(ebase + n16) * DMODEL + c * 32 + 16 * g;
    float4 w0 = *(const float4*)(wr);
    float4 w1 = *(const float4*)(wr + 4);
    float4 w2 = *(const float4*)(wr + 8);
    float4 w3 = *(const float4*)(wr + 12);
    v16h bf;
    bf[0]=(half_t)w0.x; bf[1]=(half_t)w0.y; bf[2]=(half_t)w0.z; bf[3]=(half_t)w0.w;
    bf[4]=(half_t)w1.x; bf[5]=(half_t)w1.y; bf[6]=(half_t)w1.z; bf[7]=(half_t)w1.w;
    bf[8]=(half_t)w2.x; bf[9]=(half_t)w2.y; bf[10]=(half_t)w2.z; bf[11]=(half_t)w2.w;
    bf[12]=(half_t)w3.x; bf[13]=(half_t)w3.y; bf[14]=(half_t)w3.z; bf[15]=(half_t)w3.w;

    acc = __builtin_amdgcn_wmma_f32_16x16x32_f16(false, a, false, bf,
                                                 (short)0, acc, false, false);
  }
  const float bv = bias[ebase + n16];
#pragma unroll
  for (int r = 0; r < 8; ++r) {
    const int row = rowbase + r + 8 * g;           // C layout: M = r + 8*(lane/16)
    const half_t val = (half_t)(acc[r] + bv);
    if (transposeN == 0) {
      Y[(size_t)row * DMODEL + ebase + n16] = val;
    } else {
      const int bb = row / transposeN;
      const int nn = row % transposeN;
      Y[((size_t)bb * DMODEL + ebase + n16) * (size_t)transposeN + nn] = val;
    }
  }
}

// ---------------------------------------------------------------------------
// Flash attention. Block = 256 threads = 8 waves, each wave 16 query rows.
// K tiles (64x128 f16, 16KB) double-buffered in LDS via async-to-LDS DMA.
// V read directly from pre-transposed global vT (L2-resident).
// 32 WMMAs per wave per key tile.
// ---------------------------------------------------------------------------
__global__ __launch_bounds__(256) void attn_kernel(
    const half_t* __restrict__ qh, const half_t* __restrict__ kh,
    const half_t* __restrict__ vT, float* __restrict__ out,
    int Nq, int Nk) {
  __shared__ __align__(64) half_t skh[2][KTILE * DMODEL];  // 2 x 16KB, [key][feat]
  __shared__ __align__(64) half_t sp[8][16 * KTILE];       // per-wave P, 16KB

  const int tid  = threadIdx.x;
  const int wave = tid >> 5;
  const int lane = tid & 31;
  const int n16  = lane & 15;
  const int g    = lane >> 4;

  const int qtiles = Nq / NQTILE;
  const int b  = blockIdx.x / qtiles;
  const int qt = blockIdx.x % qtiles;
  const size_t batch_off = (size_t)b * Nq * DMODEL;
  const int qbase = qt * NQTILE + wave * 16;

  const half_t* kbat = kh + batch_off;                       // saddr base
  const half_t* vTb  = vT + (size_t)b * DMODEL * Nk;

  // Preload q fragments (16 rows x 128 feats in registers).
  v16h qa[4];
  {
    const half_t* qrow = qh + batch_off + (size_t)(qbase + n16) * DMODEL;
#pragma unroll
    for (int c = 0; c < 4; ++c) {
      const half_t* p = qrow + c * 32 + 16 * g;
      v8h lo = *(const v8h*)(p);
      v8h hi = *(const v8h*)(p + 8);
#pragma unroll
      for (int h = 0; h < 8; ++h) { qa[c][h] = lo[h]; qa[c][8 + h] = hi[h]; }
    }
  }

  v8f o[8] = {};
  float m[8], l[8];
#pragma unroll
  for (int r = 0; r < 8; ++r) { m[r] = -1e30f; l[r] = 0.0f; }

  // Async DMA of one K tile into LDS buffer `buf`: 256 threads x 64B.
  auto issue_tile = [&](int kt, int buf) {
    unsigned ldsoff = (unsigned)(size_t)(&skh[buf][0]) + (unsigned)tid * 64u;
    unsigned goff   = (unsigned)kt * (KTILE * DMODEL * 2u) + (unsigned)tid * 64u;
    asm volatile(
        "global_load_async_to_lds_b128 %0, %1, %2\n\t"
        "global_load_async_to_lds_b128 %0, %1, %2 offset:16\n\t"
        "global_load_async_to_lds_b128 %0, %1, %2 offset:32\n\t"
        "global_load_async_to_lds_b128 %0, %1, %2 offset:48"
        :: "v"(ldsoff), "v"(goff), "s"(kbat)
        : "memory");
  };

  const float scale = 0.08838834764831845f;   // 1/sqrt(128)
  const int nkt = Nk / KTILE;

  issue_tile(0, 0);
  wait_async0();
  __syncthreads();

  for (int kt = 0; kt < nkt; ++kt) {
    const int buf = kt & 1;
    if (kt + 1 < nkt) issue_tile(kt + 1, buf ^ 1);   // overlap copy with compute

    // ---- S = q (16x128) @ K_tile^T : four 16x16 fragments ----
    const half_t* kt_lds = &skh[buf][0];
    v8f s[4];
#pragma unroll
    for (int j = 0; j < 4; ++j) {
      v8f acc = {};
#pragma unroll
      for (int c = 0; c < 4; ++c) {
        // B[k][n] = K[key=j*16+n][feat=c*32+16g+h], 16 halves contiguous
        const half_t* p = kt_lds + (j * 16 + n16) * DMODEL + c * 32 + 16 * g;
        v16h bf = *(const v16h*)(p);
        acc = __builtin_amdgcn_wmma_f32_16x16x32_f16(false, qa[c], false, bf,
                                                     (short)0, acc, false, false);
      }
      s[j] = acc;
    }

    // ---- online softmax: row's 16 key cols live across this lane-half ----
    float corr[8];
#pragma unroll
    for (int r = 0; r < 8; ++r) {
      float s0 = s[0][r] * scale, s1 = s[1][r] * scale;
      float s2 = s[2][r] * scale, s3 = s[3][r] * scale;
      float t = fmaxf(fmaxf(s0, s1), fmaxf(s2, s3));
      t = fmaxf(t, __shfl_xor(t, 1, 32));
      t = fmaxf(t, __shfl_xor(t, 2, 32));
      t = fmaxf(t, __shfl_xor(t, 4, 32));
      t = fmaxf(t, __shfl_xor(t, 8, 32));
      const float mnew = fmaxf(m[r], t);
      corr[r] = __expf(m[r] - mnew);
      m[r] = mnew;
      s0 = __expf(s0 - mnew); s1 = __expf(s1 - mnew);
      s2 = __expf(s2 - mnew); s3 = __expf(s3 - mnew);
      s[0][r] = s0; s[1][r] = s1; s[2][r] = s2; s[3][r] = s3;
      float rs = (s0 + s1) + (s2 + s3);
      rs += __shfl_xor(rs, 1, 32);
      rs += __shfl_xor(rs, 2, 32);
      rs += __shfl_xor(rs, 4, 32);
      rs += __shfl_xor(rs, 8, 32);
      l[r] = l[r] * corr[r] + rs;
    }
#pragma unroll
    for (int f = 0; f < 8; ++f)
#pragma unroll
      for (int r = 0; r < 8; ++r) o[f][r] *= corr[r];

    // ---- P: C-fragment -> A-fragment via per-wave LDS scratch ----
    half_t* pw = &sp[wave][0];
#pragma unroll
    for (int r = 0; r < 8; ++r)
#pragma unroll
      for (int j = 0; j < 4; ++j)
        pw[(r + 8 * g) * KTILE + j * 16 + n16] = (half_t)s[j][r];
    asm volatile("" ::: "memory");      // LDS is in-order per wave; block reordering

    v16h pa[2];
#pragma unroll
    for (int kc = 0; kc < 2; ++kc) {
      const half_t* pr = pw + n16 * KTILE + kc * 32 + 16 * g;
      v8h plo = *(const v8h*)(pr);       // k = kc*32 + 16g + 0..7
      v8h phi = *(const v8h*)(pr + 8);   // k = kc*32 + 16g + 8..15
#pragma unroll
      for (int h = 0; h < 8; ++h) { pa[kc][h] = plo[h]; pa[kc][8 + h] = phi[h]; }
    }

    // ---- O += P (16x64) @ V_tile (64x128), V straight from global vT ----
    const half_t* vbase = vTb + (size_t)kt * KTILE + 16 * g;
#pragma unroll
    for (int f = 0; f < 8; ++f) {
      const half_t* vp = vbase + (size_t)(f * 16 + n16) * Nk;
      v16h bv0 = *(const v16h*)(vp);        // keys 16g+h
      v16h bv1 = *(const v16h*)(vp + 32);   // keys 32+16g+h
      __builtin_prefetch(vp + KTILE, 0, 1); // next tile, same feature row
      o[f] = __builtin_amdgcn_wmma_f32_16x16x32_f16(false, pa[0], false, bv0,
                                                    (short)0, o[f], false, false);
      o[f] = __builtin_amdgcn_wmma_f32_16x16x32_f16(false, pa[1], false, bv1,
                                                    (short)0, o[f], false, false);
    }

    wait_async0();       // next K tile landed in LDS
    __syncthreads();     // publish to all waves; readers of old buffer done
  }

  // ---- epilogue: O / l -> fp32 out [B, Nq, 128] ----
#pragma unroll
  for (int f = 0; f < 8; ++f)
#pragma unroll
    for (int r = 0; r < 8; ++r) {
      const int row = qbase + r + 8 * g;
      out[batch_off + (size_t)row * DMODEL + f * 16 + n16] = o[f][r] / l[r];
    }
}

// ---------------------------------------------------------------------------
extern "C" void kernel_launch(void* const* d_in, const int* in_sizes, int n_in,
                              void* d_out, int out_size, void* d_ws, size_t ws_size,
                              hipStream_t stream) {
  const float* Q  = (const float*)d_in[0];
  const float* K  = (const float*)d_in[1];
  const float* V  = (const float*)d_in[2];
  const float* Wq = (const float*)d_in[3];
  const float* bq = (const float*)d_in[4];
  const float* Wk = (const float*)d_in[5];
  const float* bk = (const float*)d_in[6];
  const float* Wv = (const float*)d_in[7];
  const float* bv = (const float*)d_in[8];
  float* out = (float*)d_out;

  const int B = 4, N = 4096;
  const size_t elems = (size_t)B * N * DMODEL;
  half_t* qh = (half_t*)d_ws;        // [B*N][128]
  half_t* kh = qh + elems;           // [B*N][128]
  half_t* vT = kh + elems;           // [B][128][N]  (transposed V projection)

  const int rowTiles = (B * N) / 16;             // 1024
  proj_kernel<<<rowTiles, 256, 0, stream>>>(Q, Wq, bq, qh, 0);
  proj_kernel<<<rowTiles, 256, 0, stream>>>(K, Wk, bk, kh, 0);
  proj_kernel<<<rowTiles, 256, 0, stream>>>(V, Wv, bv, vT, N);

  const int attnBlocks = B * (N / NQTILE);       // 128
  attn_kernel<<<attnBlocks, 256, 0, stream>>>(qh, kh, vT, out, N, N);
}
